// FunnelLayer_18932215840907
// MI455X (gfx1250) — matrix operand order repaired and checked by reading, not compiled, not run on hardware
//
#include <hip/hip_runtime.h>
#include <hip/hip_bf16.h>

typedef unsigned short ushort_t;
typedef __attribute__((ext_vector_type(16))) __bf16 bf16x16;
typedef __attribute__((ext_vector_type(8)))  float  floatx8;
typedef int i32x4 __attribute__((vector_size(16)));   // matches builtin's V4i exactly

#define S_LEN 1024
#define D_HEAD 64
#define NHEAD 16
#define DMODEL 1024
#define FDIM 4096
#define BATCH 4

// gfx1250 async global->LDS copies (ASYNCcnt) if the toolchain exposes them
#if defined(__has_builtin)
#if __has_builtin(__builtin_amdgcn_global_load_async_to_lds_b128) && \
    __has_builtin(__builtin_amdgcn_s_wait_asynccnt)
#define USE_ASYNC_LDS 1
#endif
#endif
#ifndef USE_ASYNC_LDS
#define USE_ASYNC_LDS 0
#endif

#if USE_ASYNC_LDS
typedef __attribute__((address_space(1))) i32x4 as1_i32x4;
typedef __attribute__((address_space(3))) i32x4 as3_i32x4;
#endif

struct alignas(16) FragRaw { int4 lo, hi; };

static __device__ inline bf16x16 make_frag(int4 lo, int4 hi) {
  FragRaw r{lo, hi};
  return __builtin_bit_cast(bf16x16, r);
}

// A-fragment (16x32 bf16, M x K): lane L -> row (L&15); K elems {k0..k0+7} U {k0+16..k0+23}, k0=(L>=16)*8
static __device__ inline bf16x16 load_fragA(const ushort_t* p, int ld, int lane) {
  const ushort_t* q = p + (size_t)(lane & 15) * ld + ((lane >> 4) << 3);
  return make_frag(*(const int4*)q, *(const int4*)(q + 16));
}
// B-fragment (32x16 bf16, K x N): lane L -> col (L&15); K elems {k0..k0+7} U {k0+16..k0+23}, k0=(L>=16)*8
static __device__ inline bf16x16 load_fragB(const ushort_t* p, int ld, int lane) {
  const ushort_t* q = p + (size_t)(lane & 15) * ld + ((lane >> 4) << 3);
  return make_frag(*(const int4*)q, *(const int4*)(q + 16));
}
static __device__ inline floatx8 wmma_bf16(bf16x16 a, bf16x16 b, floatx8 c) {
  return __builtin_amdgcn_wmma_f32_16x16x32_bf16(false, a, false, b, (short)0, c, false, false);
}
static __device__ inline ushort_t f2bf(float f) {
  unsigned u = __float_as_uint(f);
  u += 0x7fffu + ((u >> 16) & 1u);
  return (ushort_t)(u >> 16);
}
static __device__ inline float gelu_f(float x) {
  const float k0 = 0.7978845608028654f, k1 = 0.044715f;
  float t = tanhf(k0 * (x + k1 * x * x * x));
  return 0.5f * x * (1.0f + t);
}

// ---------------------------------------------------------------- converters
__global__ void k_f32_to_bf16(const float* __restrict__ in, ushort_t* __restrict__ out, int n) {
  int i = blockIdx.x * blockDim.x + threadIdx.x;
  if (i < n) out[i] = f2bf(in[i]);
}
__global__ void k_transpose_to_bf16(const float* __restrict__ in, ushort_t* __restrict__ out, int R, int C) {
  int i = blockIdx.x * blockDim.x + threadIdx.x;
  if (i < R * C) {
    int r = i / C, c = i % C;
    out[(size_t)c * R + r] = f2bf(in[i]);
  }
}
// sinusoid table [2048 rows (last row zero pad), 1024], bf16
__global__ void k_build_rel(ushort_t* __restrict__ out) {
  int i = blockIdx.x * blockDim.x + threadIdx.x;   // 2048*1024 threads
  int p = i >> 10, k = i & 1023;
  float val = 0.f;
  if (p < 2 * S_LEN - 1) {
    float pos = (float)(p - (S_LEN - 1));
    int j = (k < 512) ? k : (k - 512);
    float invf = __expf(-(float)j * (9.210340371976184f / 512.0f)); // 10000^(-j/512)
    float a = pos * invf;
    val = (k < 512) ? sinf(a) : cosf(a);
  }
  out[i] = f2bf(val);
}

// ---------------------------------------------------------------- GEMM: C = A[M,K] * Bt[N,K]^T (bf16 in, f32 accum)
// 256 threads = 8 waves (4 Mx2 N), block tile 128x128, K-step 32, double-buffered LDS
// staged with gfx1250 async global->LDS copies when available.
enum { OUT_HEAD = 0, OUT_HEADT = 1, OUT_BF_GELU = 3, OUT_F32 = 4 };

#define GEMM_BM 128
#define GEMM_BN 128
#define GEMM_BK 32

// copy a 128x32 bf16 tile (row stride K) into LDS (row stride 32); 2 x b128 per thread
static __device__ inline void stage_tile(const ushort_t* __restrict__ g0, ushort_t* l0, int K, int tid) {
#pragma unroll
  for (int c = 0; c < 2; ++c) {
    int chunk = tid + c * 256;
    int row = chunk >> 2, col = (chunk & 3) << 3;
    const ushort_t* g = g0 + (size_t)row * K + col;
    ushort_t* l = l0 + row * GEMM_BK + col;
#if USE_ASYNC_LDS
    __builtin_amdgcn_global_load_async_to_lds_b128((as1_i32x4*)g, (as3_i32x4*)l, 0, 0);
#else
    *(int4*)l = *(const int4*)g;
#endif
  }
}

template <int MODE>
__global__ __launch_bounds__(256)
void k_gemm(const ushort_t* __restrict__ A, const ushort_t* __restrict__ Bt,
            int M, int N, int K, int Shead,
            const float* __restrict__ bias1, const float* __restrict__ bias2,
            void* __restrict__ out1p, void* __restrict__ out2p) {
  __shared__ ushort_t Abuf[2][GEMM_BM][GEMM_BK];   // 8KB per buffer
  __shared__ ushort_t Bbuf[2][GEMM_BN][GEMM_BK];
  const int tid = threadIdx.x;
  const int lane = tid & 31;
  const int warp = tid >> 5;         // 0..7
  const int wm = warp >> 1;          // 0..3 -> 32-row slice
  const int wn = warp & 1;           // 0..1 -> 64-col slice
  const int mblk = blockIdx.y * GEMM_BM;
  const int nblk = blockIdx.x * GEMM_BN;
  (void)M;

  floatx8 zero = {0, 0, 0, 0, 0, 0, 0, 0};
  floatx8 acc[2][4];
#pragma unroll
  for (int tm = 0; tm < 2; ++tm)
#pragma unroll
    for (int tn = 0; tn < 4; ++tn) acc[tm][tn] = zero;

  int buf = 0;
  stage_tile(A + (size_t)mblk * K, &Abuf[0][0][0], K, tid);
  stage_tile(Bt + (size_t)nblk * K, &Bbuf[0][0][0], K, tid);

  for (int k = 0; k < K; k += GEMM_BK) {
#if USE_ASYNC_LDS
    __builtin_amdgcn_s_wait_asynccnt(0);
#endif
    __syncthreads();
    if (k + GEMM_BK < K) {
      stage_tile(A + (size_t)mblk * K + k + GEMM_BK, &Abuf[buf ^ 1][0][0], K, tid);
      stage_tile(Bt + (size_t)nblk * K + k + GEMM_BK, &Bbuf[buf ^ 1][0][0], K, tid);
    }
    const ushort_t* Ab = &Abuf[buf][wm * 32][0];
    const ushort_t* Bb = &Bbuf[buf][wn * 64][0];
    bf16x16 af[2], bfr[4];
#pragma unroll
    for (int tm = 0; tm < 2; ++tm) af[tm] = load_fragA(Ab + tm * 16 * GEMM_BK, GEMM_BK, lane);
#pragma unroll
    for (int tn = 0; tn < 4; ++tn) bfr[tn] = load_fragB(Bb + tn * 16 * GEMM_BK, GEMM_BK, lane);
#pragma unroll
    for (int tm = 0; tm < 2; ++tm)
#pragma unroll
      for (int tn = 0; tn < 4; ++tn)
        acc[tm][tn] = wmma_bf16(af[tm], bfr[tn], acc[tm][tn]);
    buf ^= 1;
  }

  const int hf = lane >> 4;
  const int cl = lane & 15;
#pragma unroll
  for (int tm = 0; tm < 2; ++tm)
#pragma unroll
    for (int tn = 0; tn < 4; ++tn) {
      floatx8 a = acc[tm][tn];
      int col = nblk + wn * 64 + tn * 16 + cl;
      int rbase = mblk + wm * 32 + tm * 16 + hf * 8;
      float bv1 = bias1 ? bias1[col] : 0.f;
      if (MODE == OUT_HEAD) {
        ushort_t* o1 = (ushort_t*)out1p;
        ushort_t* o2 = (ushort_t*)out2p;
        int b = rbase / Shead;
        int h = col >> 6, e = col & 63;
        float bv2 = o2 ? bias2[col] : 0.f;
#pragma unroll
        for (int r = 0; r < 8; ++r) {
          int s = rbase + r - b * Shead;
          size_t dst = (((size_t)(b * NHEAD + h)) * Shead + s) * 64 + e;
          o1[dst] = f2bf(a[r] + bv1);
          if (o2) o2[dst] = f2bf(a[r] + bv2);
        }
      } else if (MODE == OUT_HEADT) {
        ushort_t* o1 = (ushort_t*)out1p;
        int b = rbase / Shead;
        int h = col >> 6, e = col & 63;
        ushort_t tmp[8];
#pragma unroll
        for (int r = 0; r < 8; ++r) tmp[r] = f2bf(a[r]);
        size_t dst = (((size_t)(b * NHEAD + h)) * 64 + e) * Shead + (rbase - b * Shead);
        int4 pk;
        pk.x = (int)(((unsigned)tmp[1] << 16) | tmp[0]);
        pk.y = (int)(((unsigned)tmp[3] << 16) | tmp[2]);
        pk.z = (int)(((unsigned)tmp[5] << 16) | tmp[4]);
        pk.w = (int)(((unsigned)tmp[7] << 16) | tmp[6]);
        *(int4*)(o1 + dst) = pk;
      } else if (MODE == OUT_BF_GELU) {
        ushort_t* o1 = (ushort_t*)out1p;
#pragma unroll
        for (int r = 0; r < 8; ++r)
          o1[(size_t)(rbase + r) * N + col] = f2bf(gelu_f(a[r] + bv1));
      } else { // OUT_F32
        float* o1 = (float*)out1p;
#pragma unroll
        for (int r = 0; r < 8; ++r)
          o1[(size_t)(rbase + r) * N + col] = a[r] + bv1;
      }
    }
}

// ---------------------------------------------------------------- attention (flash-style, rel-pos via windowed WMMA + LDS gather)
__global__ __launch_bounds__(128)
void k_attention(const ushort_t* __restrict__ Qw, const ushort_t* __restrict__ Qr,
                 const ushort_t* __restrict__ Kh, const ushort_t* __restrict__ Vt,
                 const ushort_t* __restrict__ Rh, ushort_t* __restrict__ ctxout) {
  __shared__ float Wbuf[4][2][16][32];
  __shared__ ushort_t Pbuf[4][16][32];
  const int lane = threadIdx.x & 31;
  const int warp = threadIdx.x >> 5;
  const int gid = blockIdx.x * 4 + warp;  // B*H*(S/16) = 4096 tiles
  const int b = gid >> 10;
  const int h = (gid >> 6) & 15;
  const int i0 = (gid & 63) << 4;
  const int hf = lane >> 4;
  const int cl = lane & 15;

  const ushort_t* Qw_bh = Qw + ((size_t)(b * NHEAD + h)) * S_LEN * D_HEAD;
  const ushort_t* Qr_bh = Qr + ((size_t)(b * NHEAD + h)) * S_LEN * D_HEAD;
  const ushort_t* K_bh  = Kh + ((size_t)(b * NHEAD + h)) * S_LEN * D_HEAD;
  const ushort_t* Vt_bh = Vt + ((size_t)(b * NHEAD + h)) * D_HEAD * S_LEN;
  const ushort_t* Rh_h  = Rh + (size_t)h * 2048 * D_HEAD;

  bf16x16 aw0 = load_fragA(Qw_bh + (size_t)i0 * D_HEAD + 0,  D_HEAD, lane);
  bf16x16 aw1 = load_fragA(Qw_bh + (size_t)i0 * D_HEAD + 32, D_HEAD, lane);
  bf16x16 ar0 = load_fragA(Qr_bh + (size_t)i0 * D_HEAD + 0,  D_HEAD, lane);
  bf16x16 ar1 = load_fragA(Qr_bh + (size_t)i0 * D_HEAD + 32, D_HEAD, lane);

  floatx8 zero = {0, 0, 0, 0, 0, 0, 0, 0};
  floatx8 ctx0 = zero, ctx1 = zero, ctx2 = zero, ctx3 = zero;
  float mrow[8], lrow[8];
#pragma unroll
  for (int r = 0; r < 8; ++r) { mrow[r] = -1e30f; lrow[r] = 0.f; }

  for (int j0 = 0; j0 < S_LEN; j0 += 32) {
    floatx8 s0 = zero, s1 = zero;
    floatx8 w00 = zero, w01 = zero, w10 = zero, w11 = zero;
    // content scores (Qw . K^T), two 16-col subtiles, K-dim = 64 = 2x32
    {
      bf16x16 bk0 = load_fragB(K_bh + (size_t)j0 * D_HEAD + 0,  D_HEAD, lane);
      bf16x16 bk1 = load_fragB(K_bh + (size_t)j0 * D_HEAD + 32, D_HEAD, lane);
      s0 = wmma_bf16(aw0, bk0, s0);
      s0 = wmma_bf16(aw1, bk1, s0);
      bf16x16 bk2 = load_fragB(K_bh + (size_t)(j0 + 16) * D_HEAD + 0,  D_HEAD, lane);
      bf16x16 bk3 = load_fragB(K_bh + (size_t)(j0 + 16) * D_HEAD + 32, D_HEAD, lane);
      s1 = wmma_bf16(aw0, bk2, s1);
      s1 = wmma_bf16(aw1, bk3, s1);
    }
    // position scores: Qr against 32-row R window; bd[di,dj] = W[di][di-dj+15]
    {
      const ushort_t* Rw0 = Rh_h + (size_t)(S_LEN - 1 + i0 - j0 - 15) * D_HEAD;
      w00 = wmma_bf16(ar0, load_fragB(Rw0 + 0, D_HEAD, lane), w00);
      w00 = wmma_bf16(ar1, load_fragB(Rw0 + 32, D_HEAD, lane), w00);
      w01 = wmma_bf16(ar0, load_fragB(Rw0 + 16 * D_HEAD + 0, D_HEAD, lane), w01);
      w01 = wmma_bf16(ar1, load_fragB(Rw0 + 16 * D_HEAD + 32, D_HEAD, lane), w01);
      const ushort_t* Rw1 = Rh_h + (size_t)(S_LEN - 1 + i0 - (j0 + 16) - 15) * D_HEAD;
      w10 = wmma_bf16(ar0, load_fragB(Rw1 + 0, D_HEAD, lane), w10);
      w10 = wmma_bf16(ar1, load_fragB(Rw1 + 32, D_HEAD, lane), w10);
      w11 = wmma_bf16(ar0, load_fragB(Rw1 + 16 * D_HEAD + 0, D_HEAD, lane), w11);
      w11 = wmma_bf16(ar1, load_fragB(Rw1 + 16 * D_HEAD + 32, D_HEAD, lane), w11);
    }
    __syncthreads();
#pragma unroll
    for (int r = 0; r < 8; ++r) {
      int di = r + hf * 8;
      Wbuf[warp][0][di][cl]      = w00[r];
      Wbuf[warp][0][di][16 + cl] = w01[r];
      Wbuf[warp][1][di][cl]      = w10[r];
      Wbuf[warp][1][di][16 + cl] = w11[r];
    }
    __syncthreads();
#pragma unroll
    for (int r = 0; r < 8; ++r) {
      int di = r + hf * 8;
      float sc0 = (s0[r] + Wbuf[warp][0][di][di - cl + 15]) * 0.125f;
      float sc1 = (s1[r] + Wbuf[warp][1][di][di - cl + 15]) * 0.125f;
      float t = fmaxf(sc0, sc1);
#pragma unroll
      for (int o = 1; o < 16; o <<= 1) t = fmaxf(t, __shfl_xor(t, o, 32));
      float nm = fmaxf(mrow[r], t);
      float corr = __expf(mrow[r] - nm);
      mrow[r] = nm;
      float p0 = __expf(sc0 - nm);
      float p1 = __expf(sc1 - nm);
      float ps = p0 + p1;
#pragma unroll
      for (int o = 1; o < 16; o <<= 1) ps += __shfl_xor(ps, o, 32);
      lrow[r] = lrow[r] * corr + ps;
      ctx0[r] *= corr; ctx1[r] *= corr; ctx2[r] *= corr; ctx3[r] *= corr;
      Pbuf[warp][di][cl]      = f2bf(p0);
      Pbuf[warp][di][16 + cl] = f2bf(p1);
    }
    __syncthreads();
    bf16x16 ap = load_fragA(&Pbuf[warp][0][0], 32, lane);
    ctx0 = wmma_bf16(ap, load_fragB(Vt_bh + (size_t)0 * 16 * S_LEN + j0, S_LEN, lane), ctx0);
    ctx1 = wmma_bf16(ap, load_fragB(Vt_bh + (size_t)1 * 16 * S_LEN + j0, S_LEN, lane), ctx1);
    ctx2 = wmma_bf16(ap, load_fragB(Vt_bh + (size_t)2 * 16 * S_LEN + j0, S_LEN, lane), ctx2);
    ctx3 = wmma_bf16(ap, load_fragB(Vt_bh + (size_t)3 * 16 * S_LEN + j0, S_LEN, lane), ctx3);
  }
#pragma unroll
  for (int r = 0; r < 8; ++r) {
    float inv = 1.0f / lrow[r];
    ctx0[r] *= inv; ctx1[r] *= inv; ctx2[r] *= inv; ctx3[r] *= inv;
  }
  // store ctx as bf16 [B*S, H*D] row-major for the Wo GEMM
  size_t rowbase = (size_t)b * S_LEN;
#pragma unroll
  for (int t = 0; t < 4; ++t) {
    floatx8 a = (t == 0) ? ctx0 : (t == 1) ? ctx1 : (t == 2) ? ctx2 : ctx3;
    int col = h * 64 + t * 16 + cl;
#pragma unroll
    for (int r = 0; r < 8; ++r) {
      int s = i0 + r + 8 * hf;
      ctxout[(rowbase + s) * DMODEL + col] = f2bf(a[r]);
    }
  }
}

// ---------------------------------------------------------------- residual + layernorm (one row per block)
__global__ __launch_bounds__(256)
void k_residual_ln(const float* __restrict__ resid, const float* __restrict__ mainp,
                   const float* __restrict__ g, const float* __restrict__ bb,
                   float* __restrict__ outf, ushort_t* __restrict__ outbf) {
  __shared__ float red[256];
  const int row = blockIdx.x, tid = threadIdx.x;
  const size_t base = (size_t)row * DMODEL;
  float v[4];
  float s = 0.f;
#pragma unroll
  for (int i = 0; i < 4; ++i) {
    int c = tid + 256 * i;
    v[i] = resid[base + c] + mainp[base + c];
    s += v[i];
  }
  red[tid] = s; __syncthreads();
  for (int o = 128; o > 0; o >>= 1) { if (tid < o) red[tid] += red[tid + o]; __syncthreads(); }
  float mu = red[0] * (1.0f / DMODEL);
  __syncthreads();
  float vs = 0.f;
#pragma unroll
  for (int i = 0; i < 4; ++i) { float c = v[i] - mu; vs += c * c; }
  red[tid] = vs; __syncthreads();
  for (int o = 128; o > 0; o >>= 1) { if (tid < o) red[tid] += red[tid + o]; __syncthreads(); }
  float inv = rsqrtf(red[0] * (1.0f / DMODEL) + 1e-5f);
#pragma unroll
  for (int i = 0; i < 4; ++i) {
    int c = tid + 256 * i;
    float y = (v[i] - mu) * inv * g[c] + bb[c];
    outf[base + c] = y;
    if (outbf) outbf[base + c] = f2bf(y);
  }
}

// ---------------------------------------------------------------- host
extern "C" void kernel_launch(void* const* d_in, const int* in_sizes, int n_in,
                              void* d_out, int out_size, void* d_ws, size_t ws_size,
                              hipStream_t stream) {
  (void)in_sizes; (void)n_in; (void)out_size; (void)ws_size;
  const float* q    = (const float*)d_in[0];
  const float* kv   = (const float*)d_in[1];
  const float* Wq   = (const float*)d_in[2];
  const float* Wk   = (const float*)d_in[3];
  const float* Wv   = (const float*)d_in[4];
  const float* Wr   = (const float*)d_in[5];
  const float* rwb  = (const float*)d_in[6];
  const float* rrb  = (const float*)d_in[7];
  const float* Wo   = (const float*)d_in[8];
  const float* ln1g = (const float*)d_in[9];
  const float* ln1b = (const float*)d_in[10];
  const float* W1   = (const float*)d_in[11];
  const float* b1   = (const float*)d_in[12];
  const float* W2   = (const float*)d_in[13];
  const float* b2   = (const float*)d_in[14];
  const float* ln2g = (const float*)d_in[15];
  const float* ln2b = (const float*)d_in[16];

  char* ws = (char*)d_ws;
  size_t off = 0;
  auto alloc = [&](size_t bytes) -> char* {
    char* p = ws + off;
    off += (bytes + 255) & ~(size_t)255;
    return p;
  };
  const int M = BATCH * S_LEN;  // 4096
  ushort_t* q_bf   = (ushort_t*)alloc((size_t)M * DMODEL * 2);
  ushort_t* kv_bf  = (ushort_t*)alloc((size_t)M * DMODEL * 2);
  ushort_t* rel_bf = (ushort_t*)alloc((size_t)2048 * DMODEL * 2);
  ushort_t* WqT    = (ushort_t*)alloc((size_t)DMODEL * DMODEL * 2);
  ushort_t* WkT    = (ushort_t*)alloc((size_t)DMODEL * DMODEL * 2);
  ushort_t* WvT    = (ushort_t*)alloc((size_t)DMODEL * DMODEL * 2);
  ushort_t* WrT    = (ushort_t*)alloc((size_t)DMODEL * DMODEL * 2);
  ushort_t* WoT    = (ushort_t*)alloc((size_t)DMODEL * DMODEL * 2);
  ushort_t* W1T    = (ushort_t*)alloc((size_t)FDIM * DMODEL * 2);
  ushort_t* W2T    = (ushort_t*)alloc((size_t)DMODEL * FDIM * 2);
  ushort_t* Qwbuf  = (ushort_t*)alloc((size_t)M * DMODEL * 2);
  ushort_t* Qrbuf  = (ushort_t*)alloc((size_t)M * DMODEL * 2);
  ushort_t* Khbuf  = (ushort_t*)alloc((size_t)M * DMODEL * 2);
  ushort_t* Vtbuf  = (ushort_t*)alloc((size_t)M * DMODEL * 2);
  ushort_t* Rhbuf  = (ushort_t*)alloc((size_t)NHEAD * 2048 * D_HEAD * 2);
  ushort_t* ctx_bf = (ushort_t*)alloc((size_t)M * DMODEL * 2);
  float*    attn_o = (float*)alloc((size_t)M * DMODEL * 4);
  float*    x_f32  = (float*)alloc((size_t)M * DMODEL * 4);
  ushort_t* x_bf   = (ushort_t*)alloc((size_t)M * DMODEL * 2);
  ushort_t* h_bf   = (ushort_t*)alloc((size_t)M * FDIM * 2);
  float*    y2     = (float*)alloc((size_t)M * DMODEL * 4);

  // 1) conversions / transposes / sinusoid table
  {
    int n = M * DMODEL;
    k_f32_to_bf16<<<(n + 255) / 256, 256, 0, stream>>>(q, q_bf, n);
    k_f32_to_bf16<<<(n + 255) / 256, 256, 0, stream>>>(kv, kv_bf, n);
  }
  k_transpose_to_bf16<<<(DMODEL * DMODEL + 255) / 256, 256, 0, stream>>>(Wq, WqT, DMODEL, DMODEL);
  k_transpose_to_bf16<<<(DMODEL * DMODEL + 255) / 256, 256, 0, stream>>>(Wk, WkT, DMODEL, DMODEL);
  k_transpose_to_bf16<<<(DMODEL * DMODEL + 255) / 256, 256, 0, stream>>>(Wv, WvT, DMODEL, DMODEL);
  k_transpose_to_bf16<<<(DMODEL * DMODEL + 255) / 256, 256, 0, stream>>>(Wr, WrT, DMODEL, DMODEL);
  k_transpose_to_bf16<<<(DMODEL * DMODEL + 255) / 256, 256, 0, stream>>>(Wo, WoT, DMODEL, DMODEL);
  k_transpose_to_bf16<<<(DMODEL * FDIM + 255) / 256, 256, 0, stream>>>(W1, W1T, DMODEL, FDIM);
  k_transpose_to_bf16<<<(FDIM * DMODEL + 255) / 256, 256, 0, stream>>>(W2, W2T, FDIM, DMODEL);
  k_build_rel<<<(2048 * 1024) / 256, 256, 0, stream>>>(rel_bf);

  // 2) projections (all WMMA bf16, LDS-staged)
  dim3 gProj(DMODEL / GEMM_BN, M / GEMM_BM);
  k_gemm<OUT_HEAD><<<gProj, 256, 0, stream>>>(q_bf, WqT, M, DMODEL, DMODEL, S_LEN, rwb, rrb, Qwbuf, Qrbuf);
  k_gemm<OUT_HEAD><<<gProj, 256, 0, stream>>>(kv_bf, WkT, M, DMODEL, DMODEL, S_LEN, nullptr, nullptr, Khbuf, nullptr);
  k_gemm<OUT_HEADT><<<gProj, 256, 0, stream>>>(kv_bf, WvT, M, DMODEL, DMODEL, S_LEN, nullptr, nullptr, Vtbuf, nullptr);
  dim3 gR(DMODEL / GEMM_BN, 2048 / GEMM_BM);
  k_gemm<OUT_HEAD><<<gR, 256, 0, stream>>>(rel_bf, WrT, 2048, DMODEL, DMODEL, 2048, nullptr, nullptr, Rhbuf, nullptr);

  // 3) attention
  k_attention<<<(BATCH * NHEAD * (S_LEN / 16)) / 4, 128, 0, stream>>>(Qwbuf, Qrbuf, Khbuf, Vtbuf, Rhbuf, ctx_bf);

  // 4) output projection + LN1
  k_gemm<OUT_F32><<<gProj, 256, 0, stream>>>(ctx_bf, WoT, M, DMODEL, DMODEL, S_LEN, nullptr, nullptr, attn_o, nullptr);
  k_residual_ln<<<M, 256, 0, stream>>>(q, attn_o, ln1g, ln1b, x_f32, x_bf);

  // 5) FFN + LN2
  dim3 gF1(FDIM / GEMM_BN, M / GEMM_BM);
  k_gemm<OUT_BF_GELU><<<gF1, 256, 0, stream>>>(x_bf, W1T, M, FDIM, DMODEL, S_LEN, b1, nullptr, h_bf, nullptr);
  dim3 gF2(DMODEL / GEMM_BN, M / GEMM_BM);
  k_gemm<OUT_F32><<<gF2, 256, 0, stream>>>(h_bf, W2T, M, DMODEL, FDIM, S_LEN, b2, nullptr, y2, nullptr);
  k_residual_ln<<<M, 256, 0, stream>>>(x_f32, y2, ln2g, ln2b, (float*)d_out, nullptr);
}
